// QAGNN_Message_Passing_77695958385200
// MI455X (gfx1250) — compile-verified, matrix operand order checked by reading, not run on hardware
//
#include <hip/hip_runtime.h>
#include <hip/hip_bf16.h>
#include <math.h>

// ---------------------------------------------------------------------------
// Types for CDNA5 WMMA
// ---------------------------------------------------------------------------
typedef _Float16 h16;
typedef __attribute__((ext_vector_type(16))) _Float16 v16h;
typedef __attribute__((ext_vector_type(8)))  _Float16 v8h;
typedef __attribute__((ext_vector_type(8)))  float    v8f;

#define HIDC 128
#define NNODES (128 * 200)      // 25600
#define NEDGES 400000
#define BN_EPSC 1e-5f

__device__ __forceinline__ float gelu_exact(float x) {
    return 0.5f * x * (1.0f + erff(x * 0.70710678118654752f));
}

// orderable-uint encoding for float atomic max (no NaNs expected)
__device__ __forceinline__ unsigned fenc(float f) {
    unsigned u = __float_as_uint(f);
    return (u & 0x80000000u) ? ~u : (u | 0x80000000u);
}
__device__ __forceinline__ float fdec(unsigned u) {
    return __uint_as_float((u & 0x80000000u) ? (u ^ 0x80000000u) : ~u);
}

// ---------------------------------------------------------------------------
// Small utility kernels
// ---------------------------------------------------------------------------
__global__ void k_fill_f32(float* __restrict__ p, float v, size_t n) {
    size_t i = (size_t)blockIdx.x * blockDim.x + threadIdx.x;
    if (i < n) p[i] = v;
}

__global__ void k_cvt(const float* __restrict__ s, h16* __restrict__ d, size_t n) {
    size_t i = (size_t)blockIdx.x * blockDim.x + threadIdx.x;
    if (i < n) d[i] = (h16)s[i];
}

// strided f32 -> f16 (rows x cols, separate leading dims)
__global__ void k_cvt_str(const float* __restrict__ s, h16* __restrict__ d,
                          int rows, int cols, int sld, int dld) {
    size_t i = (size_t)blockIdx.x * blockDim.x + threadIdx.x;
    if (i >= (size_t)rows * cols) return;
    int r = (int)(i / cols), c = (int)(i % cols);
    d[(size_t)r * dld + c] = (h16)s[(size_t)r * sld + c];
}

// ---------------------------------------------------------------------------
// Node embeddings: x2h[:,128:256] = [gelu(lin(onehot(nt))), gelu(lin(sin(js*s)))]
// One block (128 threads) per node.
// ---------------------------------------------------------------------------
__global__ __launch_bounds__(128)
void k_node_emb(const float* __restrict__ nscore, const int* __restrict__ ntype,
                const float* __restrict__ wnt, const float* __restrict__ bnt,   // 64x4, 64
                const float* __restrict__ wsc, const float* __restrict__ bsc,   // 64x64, 64
                h16* __restrict__ x2h) {
    __shared__ float bs[64];
    int node = blockIdx.x;
    int t = threadIdx.x;
    float s = nscore[node];
    if (t < 64) bs[t] = __sinf(__powf(1.1f, (float)t) * s);
    __syncthreads();
    if (t < 64) {
        int nt = ntype[node];
        float v = wnt[t * 4 + nt] + bnt[t];
        x2h[(size_t)node * 256 + 128 + t] = (h16)gelu_exact(v);
    } else {
        int j = t - 64;
        float a = bsc[j];
        #pragma unroll 8
        for (int i = 0; i < 64; ++i) a += wsc[j * 64 + i] * bs[i];
        x2h[(size_t)node * 256 + 192 + j] = (h16)gelu_exact(a);
    }
}

// ---------------------------------------------------------------------------
// Edge encoder first linear: one-hot input => 3 column gathers of W1 + bias,
// then BN + ReLU, emit f16.  (48-wide GEMM avoided entirely.)
// ---------------------------------------------------------------------------
__global__ void k_edge_h1(const int* __restrict__ src, const int* __restrict__ dst,
                          const int* __restrict__ etype, const int* __restrict__ ntflat,
                          const float* __restrict__ w1, const float* __restrict__ b1,
                          const float* __restrict__ bnb, const float* __restrict__ bng,
                          const float* __restrict__ bnm, const float* __restrict__ bnv,
                          h16* __restrict__ h1, int E) {
    size_t i = (size_t)blockIdx.x * blockDim.x + threadIdx.x;
    if (i >= (size_t)E * HIDC) return;
    int e = (int)(i >> 7), j = (int)(i & 127);
    int et = etype[e];
    int ht = ntflat[src[e]];
    int tt = ntflat[dst[e]];
    const float* wr = w1 + (size_t)j * 48;
    float h = wr[et] + wr[40 + ht] + wr[44 + tt] + b1[j];
    h = (h - bnm[j]) * rsqrtf(bnv[j] + BN_EPSC) * bng[j] + bnb[j];
    h1[i] = (h16)fmaxf(h, 0.0f);
}

// ---------------------------------------------------------------------------
// WMMA GEMM:  C[M x 128] = A[M x K] * W^T (+ bias, + epilogue)
// A: f16 row-major (lda), W: f16 row-major [128][ldw] (dout x din slice).
// Block = 256 threads = 8 waves; block covers 64 rows x 128 cols.
// Wave wv -> row group rg = wv>>2 (32 rows), col group cg = wv&3 (32 cols).
// Each wave register-blocks 2x2 WMMA tiles (32x32 patch): per K-step it loads
// 2 A-frags + 2 B-frags (8 x b128) and issues 4 WMMAs -> 2 loads/WMMA,
// doubling cache-level arithmetic intensity vs 1x1 tiling.
// Fragment layout per CDNA5 ISA 7.12.2 (16-bit A 16x32; symmetric B from
// row-major W; 32-bit C/D: VGPR r -> M=r / r+8 by lane half).
// ---------------------------------------------------------------------------
#define EPI_F32          0   // outF (ld 128) = acc + bias
#define EPI_F16          1   // outH (ldo)    = acc (+bias)
#define EPI_F16_ATTR     2   // outH (ldo)    = (acc+bias) * attr[row]
#define EPI_F16_BNRELU   3   // outH (ldo)    = relu(bn(acc+bias))
#define EPI_F16_GELU     4   // outH (ldo)    = gelu(acc+bias)
#define EPI_F32_ADD_GELU 5   // outF (ld 128) = gelu(acc+bias+add[row*128+col])

template <int EPI>
__global__ __launch_bounds__(256)
void k_gemm(const h16* __restrict__ A, int lda,
            const h16* __restrict__ W, int ldw, int K,
            const float* __restrict__ bias,
            float* __restrict__ outF, h16* __restrict__ outH, int ldo,
            const float* __restrict__ attr,
            const float* __restrict__ bnb, const float* __restrict__ bng,
            const float* __restrict__ bnm, const float* __restrict__ bnv,
            const float* __restrict__ addend) {
    const int lane = threadIdx.x & 31;
    const int wv   = threadIdx.x >> 5;          // 8 waves
    const int rg   = wv >> 2;                   // 0..1 -> 32-row group
    const int cg   = wv & 3;                    // 0..3 -> 32-col group
    const int m0   = blockIdx.x * 64;
    const int half = lane >> 4;                 // 0 or 1
    const int l15  = lane & 15;

    const h16* arow0 = A + (size_t)(m0 + rg * 32 + l15) * lda + half * 8;
    const h16* arow1 = arow0 + (size_t)16 * lda;
    const h16* wrow0 = W + (size_t)(cg * 32 + l15) * ldw + half * 8;
    const h16* wrow1 = wrow0 + (size_t)16 * ldw;

    v8f acc[2][2] = {};
    for (int kk = 0; kk < K; kk += 32) {
        v16h a0, a1, b0, b1;
        *((v8h*)&a0)       = *(const v8h*)(arow0 + kk);
        *(((v8h*)&a0) + 1) = *(const v8h*)(arow0 + kk + 16);
        *((v8h*)&a1)       = *(const v8h*)(arow1 + kk);
        *(((v8h*)&a1) + 1) = *(const v8h*)(arow1 + kk + 16);
        *((v8h*)&b0)       = *(const v8h*)(wrow0 + kk);
        *(((v8h*)&b0) + 1) = *(const v8h*)(wrow0 + kk + 16);
        *((v8h*)&b1)       = *(const v8h*)(wrow1 + kk);
        *(((v8h*)&b1) + 1) = *(const v8h*)(wrow1 + kk + 16);
        acc[0][0] = __builtin_amdgcn_wmma_f32_16x16x32_f16(
                        false, a0, false, b0, (short)0, acc[0][0], false, false);
        acc[0][1] = __builtin_amdgcn_wmma_f32_16x16x32_f16(
                        false, a0, false, b1, (short)0, acc[0][1], false, false);
        acc[1][0] = __builtin_amdgcn_wmma_f32_16x16x32_f16(
                        false, a1, false, b0, (short)0, acc[1][0], false, false);
        acc[1][1] = __builtin_amdgcn_wmma_f32_16x16x32_f16(
                        false, a1, false, b1, (short)0, acc[1][1], false, false);
    }

    #pragma unroll
    for (int i = 0; i < 2; ++i) {
        #pragma unroll
        for (int j = 0; j < 2; ++j) {
            #pragma unroll
            for (int r = 0; r < 8; ++r) {
                int row = m0 + rg * 32 + i * 16 + r + half * 8;
                int col = cg * 32 + j * 16 + l15;
                float v = acc[i][j][r] + (bias ? bias[col] : 0.0f);
                if (EPI == EPI_F32) {
                    outF[(size_t)row * HIDC + col] = v;
                } else if (EPI == EPI_F16) {
                    outH[(size_t)row * ldo + col] = (h16)v;
                } else if (EPI == EPI_F16_ATTR) {
                    outH[(size_t)row * ldo + col] = (h16)(v * attr[row]);
                } else if (EPI == EPI_F16_BNRELU) {
                    v = (v - bnm[col]) * rsqrtf(bnv[col] + BN_EPSC) * bng[col] + bnb[col];
                    outH[(size_t)row * ldo + col] = (h16)fmaxf(v, 0.0f);
                } else if (EPI == EPI_F16_GELU) {
                    outH[(size_t)row * ldo + col] = (h16)gelu_exact(v);
                } else { // EPI_F32_ADD_GELU
                    v += addend[(size_t)row * HIDC + col];
                    outF[(size_t)row * HIDC + col] = gelu_exact(v);
                }
            }
        }
    }
}

// ---------------------------------------------------------------------------
// Attention / segment-softmax kernels
// ---------------------------------------------------------------------------
__global__ void k_deg(const int* __restrict__ src, float* __restrict__ deg, int E) {
    int e = blockIdx.x * blockDim.x + threadIdx.x;
    if (e < E) atomicAdd(&deg[src[e]], 1.0f);
}

// scores[e][h] = sum_k (Qx[src]*inv) * (Kx[dst] + Ke[e]);  atomic max into smax[src]
__global__ void k_scores(const int* __restrict__ src, const int* __restrict__ dst,
                         const float* __restrict__ Qx, const float* __restrict__ Kx,
                         const h16* __restrict__ Ke,
                         float* __restrict__ scores, unsigned* __restrict__ smax, int E) {
    int e = blockIdx.x * blockDim.x + threadIdx.x;
    if (e >= E) return;
    const float inv = 0.17677669529663688f; // 1/sqrt(32)
    int s = src[e], d = dst[e];
    const float* q = Qx + (size_t)s * HIDC;
    const float* k = Kx + (size_t)d * HIDC;
    const h16*  ke = Ke + (size_t)e * HIDC;
    float sc[4] = {0.f, 0.f, 0.f, 0.f};
    #pragma unroll 4
    for (int i = 0; i < HIDC; ++i) {
        float kv = k[i] + (float)ke[i];
        sc[i >> 5] += (q[i] * inv) * kv;
    }
    #pragma unroll
    for (int h = 0; h < 4; ++h) {
        scores[(size_t)e * 4 + h] = sc[h];
        atomicMax(&smax[(size_t)s * 4 + h], fenc(sc[h]));
    }
}

// ex = exp(score - smax[src]);  atomic add into ssum[src]
__global__ void k_expsum(const int* __restrict__ src,
                         float* __restrict__ scores, const unsigned* __restrict__ smax,
                         float* __restrict__ ssum, int E) {
    size_t i = (size_t)blockIdx.x * blockDim.x + threadIdx.x;
    if (i >= (size_t)E * 4) return;
    int e = (int)(i >> 2), h = (int)(i & 3);
    int s = src[e];
    float m = fdec(smax[(size_t)s * 4 + h]);
    float ex = __expf(scores[i] - m);
    scores[i] = ex;
    atomicAdd(&ssum[(size_t)s * 4 + h], ex);
}

// aggr[dst] += (Mx[src] + Me[e]) * alpha,  alpha = ex/(ssum+eps) * deg[src]
__global__ void k_aggr(const int* __restrict__ src, const int* __restrict__ dst,
                       const float* __restrict__ scores, const float* __restrict__ ssum,
                       const float* __restrict__ deg,
                       const float* __restrict__ Mx, const h16* __restrict__ Me,
                       float* __restrict__ aggr, int E) {
    size_t i = (size_t)blockIdx.x * blockDim.x + threadIdx.x;
    if (i >= (size_t)E * HIDC) return;
    int e = (int)(i >> 7), k = (int)(i & 127), h = k >> 5;
    int s = src[e], d = dst[e];
    float alpha = scores[(size_t)e * 4 + h] /
                  (ssum[(size_t)s * 4 + h] + 1e-16f) * deg[s];
    float m = Mx[(size_t)s * HIDC + k] + (float)Me[i];
    atomicAdd(&aggr[(size_t)d * HIDC + k], m * alpha);
}

// ---------------------------------------------------------------------------
// Host-side orchestration
// ---------------------------------------------------------------------------
static inline int cdiv(long long a, long long b) { return (int)((a + b - 1) / b); }

extern "C" void kernel_launch(void* const* d_in, const int* in_sizes, int n_in,
                              void* d_out, int out_size, void* d_ws, size_t ws_size,
                              hipStream_t stream) {
    (void)in_sizes; (void)n_in; (void)out_size; (void)ws_size;
    const int Nn = NNODES, E = NEDGES;

    // --- input indexing: top-level in setup_inputs() order; params pytree
    //     flattened with sorted dict keys (jax tree_leaves semantics) ---
    const float* H      = (const float*)d_in[0];
    const float* eattr  = (const float*)d_in[1];
    const float* nscore = (const float*)d_in[2];
    int p = 3;
    const float* Vh_b = (const float*)d_in[p++]; const float* Vh_w = (const float*)d_in[p++];
    const float* Vx_b = (const float*)d_in[p++]; const float* Vx_w = (const float*)d_in[p++];
    const float* ee_bn_beta = (const float*)d_in[p++]; const float* ee_bn_gamma = (const float*)d_in[p++];
    const float* ee_bn_mean = (const float*)d_in[p++]; const float* ee_bn_var   = (const float*)d_in[p++];
    const float* ee_l1_b = (const float*)d_in[p++]; const float* ee_l1_w = (const float*)d_in[p++];
    const float* ee_l2_b = (const float*)d_in[p++]; const float* ee_l2_w = (const float*)d_in[p++];
    const float* ent_b = (const float*)d_in[p++]; const float* ent_w = (const float*)d_in[p++];
    const float* esc_b = (const float*)d_in[p++]; const float* esc_w = (const float*)d_in[p++];
    // per-layer (sorted): key, mlp1, mlp2, mlp_bn(beta,gamma,mean,var), msg, query
    const float *Lkey_b[2], *Lkey_w[2], *Lm1_b[2], *Lm1_w[2], *Lm2_b[2], *Lm2_w[2];
    const float *Lbn_b[2], *Lbn_g[2], *Lbn_m[2], *Lbn_v[2];
    const float *Lmsg_b[2], *Lmsg_w[2], *Lq_b[2], *Lq_w[2];
    for (int l = 0; l < 2; ++l) {
        Lkey_b[l] = (const float*)d_in[p++]; Lkey_w[l] = (const float*)d_in[p++];
        Lm1_b[l]  = (const float*)d_in[p++]; Lm1_w[l]  = (const float*)d_in[p++];
        Lm2_b[l]  = (const float*)d_in[p++]; Lm2_w[l]  = (const float*)d_in[p++];
        Lbn_b[l]  = (const float*)d_in[p++]; Lbn_g[l]  = (const float*)d_in[p++];
        Lbn_m[l]  = (const float*)d_in[p++]; Lbn_v[l]  = (const float*)d_in[p++];
        Lmsg_b[l] = (const float*)d_in[p++]; Lmsg_w[l] = (const float*)d_in[p++];
        Lq_b[l]   = (const float*)d_in[p++]; Lq_w[l]   = (const float*)d_in[p++];
    }
    const int* eidx  = (const int*)d_in[p++];        // (2,E): src then dst
    const int* etype = (const int*)d_in[p++];
    const int* ntype = (const int*)d_in[p++];
    const int* esrc = eidx;
    const int* edst = eidx + E;

    // --- workspace carve ---
    char* wp = (char*)d_ws;
    auto alloc = [&](size_t bytes) -> void* {
        void* q = (void*)wp; wp += (bytes + 255) & ~(size_t)255; return q;
    };
    h16*  x2h   = (h16*)  alloc((size_t)Nn * 256 * 2);   // [X | nfe] f16
    h16*  Hh    = (h16*)  alloc((size_t)Nn * HIDC * 2);  // original H f16
    h16*  h1    = (h16*)  alloc((size_t)E * HIDC * 2);   // edge hidden; reused as Ke
    h16*  eemb  = (h16*)  alloc((size_t)E * HIDC * 2);   // edge_emb f16
    h16*  Me    = (h16*)  alloc((size_t)E * HIDC * 2);
    float* Kx   = (float*)alloc((size_t)Nn * HIDC * 4);
    float* Mx   = (float*)alloc((size_t)Nn * HIDC * 4);
    float* Qx   = (float*)alloc((size_t)Nn * HIDC * 4);
    float* scores = (float*)alloc((size_t)E * 4 * 4);
    unsigned* smax = (unsigned*)alloc((size_t)Nn * 4 * 4);
    float* ssum = (float*)alloc((size_t)Nn * 4 * 4);
    float* deg  = (float*)alloc((size_t)Nn * 4);
    float* aggr = (float*)alloc((size_t)Nn * HIDC * 4);
    h16*  aggrh = (h16*)  alloc((size_t)Nn * HIDC * 2);
    h16*  hbuf  = (h16*)  alloc((size_t)Nn * HIDC * 2);
    float* vtmp = (float*)alloc((size_t)Nn * HIDC * 4);
    h16* keywh[2], *msgwh[2], *qrywh[2], *m1wh[2], *m2wh[2];
    for (int l = 0; l < 2; ++l) {
        keywh[l] = (h16*)alloc(128 * 384 * 2);
        msgwh[l] = (h16*)alloc(128 * 384 * 2);
        qrywh[l] = (h16*)alloc(128 * 256 * 2);
        m1wh[l]  = (h16*)alloc(128 * 128 * 2);
        m2wh[l]  = (h16*)alloc(128 * 128 * 2);
    }
    h16* l2wh = (h16*)alloc(128 * 128 * 2);
    h16* Vhwh = (h16*)alloc(128 * 128 * 2);
    h16* Vxwh = (h16*)alloc(128 * 128 * 2);
    h16* Ke = h1;  // h1 dead after edge-encoder GEMM

    const int T = 256;
    // --- weight + H conversions ---
    k_cvt<<<cdiv(128 * 128, T), T, 0, stream>>>(ee_l2_w, l2wh, 128 * 128);
    k_cvt<<<cdiv(128 * 128, T), T, 0, stream>>>(Vh_w, Vhwh, 128 * 128);
    k_cvt<<<cdiv(128 * 128, T), T, 0, stream>>>(Vx_w, Vxwh, 128 * 128);
    for (int l = 0; l < 2; ++l) {
        k_cvt<<<cdiv(128 * 384, T), T, 0, stream>>>(Lkey_w[l], keywh[l], 128 * 384);
        k_cvt<<<cdiv(128 * 384, T), T, 0, stream>>>(Lmsg_w[l], msgwh[l], 128 * 384);
        k_cvt<<<cdiv(128 * 256, T), T, 0, stream>>>(Lq_w[l],   qrywh[l], 128 * 256);
        k_cvt<<<cdiv(128 * 128, T), T, 0, stream>>>(Lm1_w[l],  m1wh[l],  128 * 128);
        k_cvt<<<cdiv(128 * 128, T), T, 0, stream>>>(Lm2_w[l],  m2wh[l],  128 * 128);
    }
    k_cvt<<<cdiv((long long)Nn * HIDC, T), T, 0, stream>>>(H, Hh, (size_t)Nn * HIDC);
    k_cvt_str<<<cdiv((long long)Nn * HIDC, T), T, 0, stream>>>(H, x2h, Nn, HIDC, HIDC, 256);

    // --- node-feature embeddings (second half of x2h) ---
    k_node_emb<<<Nn, 128, 0, stream>>>(nscore, ntype, ent_w, ent_b, esc_w, esc_b, x2h);

    // --- edge encoder ---
    k_edge_h1<<<cdiv((long long)E * HIDC, T), T, 0, stream>>>(
        esrc, edst, etype, ntype, ee_l1_w, ee_l1_b,
        ee_bn_beta, ee_bn_gamma, ee_bn_mean, ee_bn_var, h1, E);
    k_gemm<EPI_F16_ATTR><<<E / 64, 256, 0, stream>>>(
        h1, HIDC, l2wh, HIDC, HIDC, ee_l2_b,
        nullptr, eemb, HIDC, eattr, nullptr, nullptr, nullptr, nullptr, nullptr);

    // --- per-edge source out-degree (constant across layers) ---
    k_fill_f32<<<cdiv(Nn, T), T, 0, stream>>>(deg, 0.0f, (size_t)Nn);
    k_deg<<<cdiv(E, T), T, 0, stream>>>(esrc, deg, E);

    // --- GNN layers ---
    for (int l = 0; l < 2; ++l) {
        // node-side linears over x2 (bias applied here)
        k_gemm<EPI_F32><<<Nn / 64, 256, 0, stream>>>(
            x2h, 256, keywh[l], 384, 256, Lkey_b[l],
            Kx, nullptr, 0, nullptr, nullptr, nullptr, nullptr, nullptr, nullptr);
        k_gemm<EPI_F32><<<Nn / 64, 256, 0, stream>>>(
            x2h, 256, msgwh[l], 384, 256, Lmsg_b[l],
            Mx, nullptr, 0, nullptr, nullptr, nullptr, nullptr, nullptr, nullptr);
        k_gemm<EPI_F32><<<Nn / 64, 256, 0, stream>>>(
            x2h, 256, qrywh[l], 256, 256, Lq_b[l],
            Qx, nullptr, 0, nullptr, nullptr, nullptr, nullptr, nullptr, nullptr);
        // edge-side linears over edge_emb (weight columns 256..383, no bias)
        k_gemm<EPI_F16><<<E / 64, 256, 0, stream>>>(
            eemb, HIDC, keywh[l] + 256, 384, HIDC, nullptr,
            nullptr, Ke, HIDC, nullptr, nullptr, nullptr, nullptr, nullptr, nullptr);
        k_gemm<EPI_F16><<<E / 64, 256, 0, stream>>>(
            eemb, HIDC, msgwh[l] + 256, 384, HIDC, nullptr,
            nullptr, Me, HIDC, nullptr, nullptr, nullptr, nullptr, nullptr, nullptr);
        // segment softmax over src
        k_fill_f32<<<cdiv((long long)Nn * 4, T), T, 0, stream>>>((float*)smax, 0.0f, (size_t)Nn * 4);
        k_fill_f32<<<cdiv((long long)Nn * 4, T), T, 0, stream>>>(ssum, 0.0f, (size_t)Nn * 4);
        k_scores<<<cdiv(E, T), T, 0, stream>>>(esrc, edst, Qx, Kx, Ke, scores, smax, E);
        k_expsum<<<cdiv((long long)E * 4, T), T, 0, stream>>>(esrc, scores, smax, ssum, E);
        // weighted scatter-add at dst
        k_fill_f32<<<cdiv((long long)Nn * HIDC, T), T, 0, stream>>>(aggr, 0.0f, (size_t)Nn * HIDC);
        k_aggr<<<cdiv((long long)E * HIDC, T), T, 0, stream>>>(
            esrc, edst, scores, ssum, deg, Mx, Me, aggr, E);
        // node MLP: mlp2(relu(bn(mlp1(aggr)))) -> gelu -> X (f16 half of x2h)
        k_cvt<<<cdiv((long long)Nn * HIDC, T), T, 0, stream>>>(aggr, aggrh, (size_t)Nn * HIDC);
        k_gemm<EPI_F16_BNRELU><<<Nn / 64, 256, 0, stream>>>(
            aggrh, HIDC, m1wh[l], HIDC, HIDC, Lm1_b[l],
            nullptr, hbuf, HIDC, nullptr, Lbn_b[l], Lbn_g[l], Lbn_m[l], Lbn_v[l], nullptr);
        k_gemm<EPI_F16_GELU><<<Nn / 64, 256, 0, stream>>>(
            hbuf, HIDC, m2wh[l], HIDC, HIDC, Lm2_b[l],
            nullptr, x2h, 256, nullptr, nullptr, nullptr, nullptr, nullptr, nullptr);
    }

    // --- output: gelu(Vh(H) + Vx(X)) ---
    k_gemm<EPI_F32><<<Nn / 64, 256, 0, stream>>>(
        Hh, HIDC, Vhwh, HIDC, HIDC, Vh_b,
        vtmp, nullptr, 0, nullptr, nullptr, nullptr, nullptr, nullptr, nullptr);
    k_gemm<EPI_F32_ADD_GELU><<<Nn / 64, 256, 0, stream>>>(
        x2h, 256, Vxwh, HIDC, HIDC, Vx_b,
        (float*)d_out, nullptr, 0, nullptr, nullptr, nullptr, nullptr, nullptr, vtmp);
}
